// NNPytorchHessian_3659312136357
// MI455X (gfx1250) — matrix operand order repaired and checked by reading, not compiled, not run on hardware
//
#include <hip/hip_runtime.h>
#include <hip/hip_bf16.h>

typedef __attribute__((ext_vector_type(8)))  _Float16 v8h;
typedef __attribute__((ext_vector_type(16))) _Float16 v16h;
typedef __attribute__((ext_vector_type(8)))  float    v8f;

// Network dims (fixed by the reference): N=2048 examples, D=128 inputs, H=256 hidden.

struct Smem {
  _Float16 Bmat[128 * 256];  // B = (W1*Diag(t1)) @ W2, f16 row-major (64 KB)
  _Float16 sh[512];          // [s1 | s2] in f16 (K-axis scale for H-GEMM)
  _Float16 t1h[256];         // t1 in f16 (K-axis scale for B-GEMM)
  float xs[128];
  float a1[256];
  float t1[256];
  float vv[256];             // v = t2 * W3
  float g1[256];
  float fsum;
};

__device__ __forceinline__ v8h ld8h(const _Float16* p) {
  return *reinterpret_cast<const v8h*>(p);
}
__device__ __forceinline__ v16h cat16(v8h a, v8h b) {
  return __builtin_shufflevector(a, b, 0,1,2,3,4,5,6,7,8,9,10,11,12,13,14,15);
}
__device__ __forceinline__ v8f wmma_f16(v16h a, v16h b, v8f c) {
  return __builtin_amdgcn_wmma_f32_16x16x32_f16(false, a, false, b, (short)0, c,
                                                false, false);
}

// ---- prep: f32 -> f16 weight staging (shared across all examples) ----
__global__ void prep_w1h(const float* __restrict__ W1, _Float16* __restrict__ W1h) {
  int i = blockIdx.x * blockDim.x + threadIdx.x;
  if (i < 128 * 256) W1h[i] = (_Float16)W1[i];
}
__global__ void prep_w2t(const float* __restrict__ W2, _Float16* __restrict__ W2Th) {
  int i = blockIdx.x * blockDim.x + threadIdx.x;
  if (i < 256 * 256) {
    int k = i >> 8, j = i & 255;            // W2Th[k][j] = W2[j][k]
    W2Th[i] = (_Float16)W2[j * 256 + k];
  }
}

// ---- main: one block per example, f + df + Hessian ----
__global__ void __launch_bounds__(256)
hessian_mlp_kernel(const float* __restrict__ x,  const float* __restrict__ W1,
                   const float* __restrict__ b1, const float* __restrict__ W2,
                   const float* __restrict__ b2, const float* __restrict__ W3,
                   const float* __restrict__ b3,
                   const _Float16* __restrict__ W1h, const _Float16* __restrict__ W2Th,
                   float* __restrict__ out_f, float* __restrict__ out_df,
                   float* __restrict__ out_d2f)
{
  extern __shared__ char smraw[];
  Smem& sm = *reinterpret_cast<Smem*>(smraw);
  const int tid = threadIdx.x;
  const int n = blockIdx.x;

  if (tid < 128) sm.xs[tid] = x[(size_t)n * 128 + tid];
  if (tid == 0) sm.fsum = b3[0];
  __syncthreads();

  // ---- layer 1: z1 = x@W1 + b1 (thread j owns column j, coalesced over j) ----
  {
    float acc = b1[tid];
#pragma unroll 4
    for (int i = 0; i < 128; ++i) acc += sm.xs[i] * W1[i * 256 + tid];
    float a = tanhf(acc);
    float t = 1.0f - a * a;
    sm.a1[tid] = a;
    sm.t1[tid] = t;
    sm.t1h[tid] = (_Float16)t;
  }
  __syncthreads();

  // ---- layer 2: z2 = a1@W2 + b2; v = t2*W3; s2 = -2*a2*t2*W3; f partial ----
  {
    float acc = b2[tid];
#pragma unroll 4
    for (int j = 0; j < 256; ++j) acc += sm.a1[j] * W2[j * 256 + tid];
    float a = tanhf(acc);
    float t = 1.0f - a * a;
    float w3 = W3[tid];
    sm.vv[tid] = t * w3;
    sm.sh[256 + tid] = (_Float16)(-2.0f * a * t * w3);
    atomicAdd(&sm.fsum, a * w3);
  }
  __syncthreads();

  // ---- u = W2 @ v (via W2^T, coalesced); g1 = t1*u; s1 = -2*a1*t1*u ----
  {
    float acc = 0.0f;
#pragma unroll 4
    for (int k = 0; k < 256; ++k) acc += (float)W2Th[k * 256 + tid] * sm.vv[k];
    float t = sm.t1[tid];
    float a = sm.a1[tid];
    sm.g1[tid] = t * acc;
    sm.sh[tid] = (_Float16)(-2.0f * a * t * acc);
  }
  if (tid == 0) out_f[n] = sm.fsum;
  __syncthreads();

  // ---- gradient: df = W1 @ g1 ----
  if (tid < 128) {
    float acc = 0.0f;
#pragma unroll 4
    for (int j = 0; j < 256; ++j) acc += W1[tid * 256 + j] * sm.g1[j];
    out_df[(size_t)n * 128 + tid] = acc;
  }

  // ---- WMMA lane decomposition ----
  const int lane = tid & 31;
  const int wv   = tid >> 5;    // 8 waves -> 8 row-strips of 16
  const int lm   = lane & 15;
  const int lh   = lane >> 4;
  const int m0   = wv * 16;

  // ======== GEMM1: B[128x256] = (W1 Diag(t1)) @ W2, M=128 N=256 K=256 ========
  // A-fragment: row (m0+lm) of W1h, K chunks [koff,koff+8) and [koff+16,koff+24),
  // scaled on the K axis by t1h.  B-fragment: W2Th row (n0+lm) = column of W2,
  // K chunks [kb,kb+8) and [kb+16,kb+24), same K order as A.
  v16h afr[8];
  {
    const _Float16* arow = W1h + (m0 + lm) * 256;
#pragma unroll
    for (int kk = 0; kk < 8; ++kk) {
      const int koff = kk * 32 + lh * 8;
      v8h lo = ld8h(arow + koff)      * ld8h(sm.t1h + koff);
      v8h hi = ld8h(arow + koff + 16) * ld8h(sm.t1h + koff + 16);
      afr[kk] = cat16(lo, hi);
    }
  }
#pragma unroll 1
  for (int nt = 0; nt < 16; ++nt) {
    const int n0 = nt * 16;
    const _Float16* brow = W2Th + (n0 + lm) * 256;
    v8f acc = {};
#pragma unroll
    for (int kk = 0; kk < 8; ++kk) {
      const int kb = kk * 32 + lh * 8;
      acc = wmma_f16(afr[kk], cat16(ld8h(brow + kb), ld8h(brow + kb + 16)), acc);
    }
#pragma unroll
    for (int r = 0; r < 8; ++r)
      sm.Bmat[(m0 + r + 8 * lh) * 256 + n0 + lm] = (_Float16)acc[r];
  }
  __syncthreads();

  // ======== GEMM2: H[128x128] = M Diag(s) M^T,  M=[W1|B], K=512 ========
  // A-fragment: unscaled rows of M (W1h for K<256, Bmat for K>=256).
  // B-fragment: Q[K][i] = s[K]*M[i][K] -> contiguous row chunk of M times sh.
  v16h pfr[16];
  {
    const _Float16* w1row = W1h    + (m0 + lm) * 256;
    const _Float16* bro   = sm.Bmat + (m0 + lm) * 256;
#pragma unroll
    for (int kk = 0; kk < 16; ++kk) {
      const int koff = (kk & 7) * 32 + lh * 8;
      const _Float16* row = (kk < 8) ? w1row : bro;
      pfr[kk] = cat16(ld8h(row + koff), ld8h(row + koff + 16));
    }
  }
  float* Hn = out_d2f + (size_t)n * (128 * 128);
#pragma unroll 1
  for (int nt = 0; nt < 8; ++nt) {
    const int n0 = nt * 16;
    const _Float16* w1row = W1h    + (n0 + lm) * 256;
    const _Float16* bro   = sm.Bmat + (n0 + lm) * 256;
    v8f acc = {};
#pragma unroll
    for (int kk = 0; kk < 16; ++kk) {
      const int kabs = kk * 32 + lh * 8;        // absolute K in [0,512)
      const int koff = (kk & 7) * 32 + lh * 8;  // offset within source row
      const _Float16* row = (kk < 8) ? w1row : bro;
      v8h blo = ld8h(row + koff)      * ld8h(sm.sh + kabs);
      v8h bhi = ld8h(row + koff + 16) * ld8h(sm.sh + kabs + 16);
      acc = wmma_f16(pfr[kk], cat16(blo, bhi), acc);
    }
#pragma unroll
    for (int r = 0; r < 8; ++r)
      Hn[(m0 + r + 8 * lh) * 128 + n0 + lm] = acc[r];
  }
}

extern "C" void kernel_launch(void* const* d_in, const int* in_sizes, int n_in,
                              void* d_out, int out_size, void* d_ws, size_t ws_size,
                              hipStream_t stream) {
  const float* x  = (const float*)d_in[0];
  const float* W1 = (const float*)d_in[1];
  const float* b1 = (const float*)d_in[2];
  const float* W2 = (const float*)d_in[3];
  const float* b2 = (const float*)d_in[4];
  const float* W3 = (const float*)d_in[5];
  const float* b3 = (const float*)d_in[6];
  const int nex = in_sizes[0] / 128;   // 2048

  _Float16* W1h  = (_Float16*)d_ws;                                   // 64 KB
  _Float16* W2Th = (_Float16*)((char*)d_ws + 128 * 256 * sizeof(_Float16)); // 128 KB

  prep_w1h<<<(128 * 256 + 255) / 256, 256, 0, stream>>>(W1, W1h);
  prep_w2t<<<(256 * 256 + 255) / 256, 256, 0, stream>>>(W2, W2Th);

  float* out     = (float*)d_out;
  float* out_f   = out;                          // [nex]
  float* out_df  = out + nex;                    // [nex][128]
  float* out_d2f = out_df + (size_t)nex * 128;   // [nex][128][128]

  hessian_mlp_kernel<<<nex, 256, sizeof(Smem), stream>>>(
      x, W1, b1, W2, b2, W3, b3, W1h, W2Th, out_f, out_df, out_d2f);
}
